// Attention_76948634075216
// MI455X (gfx1250) — matrix operand order repaired and checked by reading, not compiled, not run on hardware
//
#include <hip/hip_runtime.h>

typedef _Float16 v8h  __attribute__((ext_vector_type(8)));
typedef _Float16 v16h __attribute__((ext_vector_type(16)));
typedef float    v8f  __attribute__((ext_vector_type(8)));

#define S_LEN 4096
#define D_DIM 64
#define QT 256           // queries per block: 8 waves x 32 queries
#define KT 64            // keys per LDS chunk
#define LSTR 80          // LDS row stride in halves (160B => 16B/32B aligned frags)
#define NTHREADS 256

// Load one 16x32 f16 A-fragment row from an LDS row pointer.
// Layout: lanes 0-15 hold K=0..15; lanes 16-31 hold K=16..31 (contiguous).
__device__ __forceinline__ v16h load_a_frag(const _Float16* row, int hi) {
    const int kb = hi * 16;
    v8h lo = *(const v8h*)(row + kb);
    v8h hc = *(const v8h*)(row + kb + 8);
    v16h r;
#pragma unroll
    for (int i = 0; i < 8; ++i) { r[i] = lo[i]; r[i + 8] = hc[i]; }
    return r;
}

__global__ __launch_bounds__(NTHREADS)
void fa_fwd_kernel(const float* __restrict__ Kg, const float* __restrict__ Vg,
                   const float* __restrict__ Qg, float* __restrict__ Og) {
    __shared__ __align__(16) _Float16 sK[KT * LSTR];     // [key][d]  (also Q staging scratch)
    __shared__ __align__(16) _Float16 sV[D_DIM * LSTR];  // [d][key]

    const int b    = blockIdx.y;
    const int q0   = blockIdx.x * QT;
    const int tid  = threadIdx.x;
    const int lane = tid & 31;
    const int wid  = tid >> 5;
    const int col  = lane & 15;  // N index (query / matrix column)
    const int hi   = lane >> 4;  // half-wave select

    const float* Kb = Kg + (size_t)b * D_DIM * S_LEN;
    const float* Vb = Vg + (size_t)b * D_DIM * S_LEN;
    const float* Qb = Qg + (size_t)b * D_DIM * S_LEN;

    // ---- Stage Q B-fragments through sK (4 passes of 64 queries), scaled 1/sqrt(D).
    // Wave wid owns local queries 32*wid..32*wid+31 as two 16-wide sub-tiles.
    // B layout: lanes 0-15 hold K(d)=0..15 contiguous, lanes 16-31 hold K(d)=16..31.
    v16h qf[2][2];  // [qtile][d-half]
    for (int p = 0; p < 4; ++p) {
#pragma unroll
        for (int e = tid; e < 64 * D_DIM; e += NTHREADS) {
            const int d = e >> 6, qq = e & 63;
            sK[qq * LSTR + d] =
                (_Float16)(Qb[(size_t)d * S_LEN + q0 + p * 64 + qq] * 0.125f);
        }
        __syncthreads();
        if ((wid >> 1) == p) {
            const int qloc = (wid & 1) * 32;
#pragma unroll
            for (int t = 0; t < 2; ++t) {
                const _Float16* qrow = sK + (qloc + t * 16 + col) * LSTR + hi * 16;
                qf[t][0] = *(const v16h*)(qrow);
                qf[t][1] = *(const v16h*)(qrow + 32);
            }
        }
        __syncthreads();
    }

    v8f   o[2][4] = {};                    // per q-tile: 4 d-chunk accumulators
    float m[2] = {-3.0e38f, -3.0e38f};     // running max per q-tile column
    float l[2] = {0.0f, 0.0f};             // running sum per q-tile column

    for (int k0 = 0; k0 < S_LEN; k0 += KT) {
        __syncthreads();  // previous chunk's readers done before overwrite
        // ---- Stage K (transposed to [key][d]) and V ([d][key]) chunks as f16
#pragma unroll
        for (int e = tid; e < KT * D_DIM; e += NTHREADS) {
            const int d = e >> 6, kk = e & 63;
            const size_t g = (size_t)d * S_LEN + k0 + kk;
            sK[kk * LSTR + d] = (_Float16)Kb[g];
            sV[d * LSTR + kk] = (_Float16)Vb[g];
        }
        // Prefetch next chunk (256B rows => two 128B points per row)
        if (k0 + KT < S_LEN) {
            const int d = tid >> 1, off = (tid & 1) * 32;
            if (tid < 128)
                __builtin_prefetch(Kb + (size_t)d * S_LEN + k0 + KT + off, 0, 0);
            else
                __builtin_prefetch(Vb + (size_t)(d - 64) * S_LEN + k0 + KT + off, 0, 0);
        }
        __syncthreads();

#pragma unroll
        for (int st = 0; st < 2; ++st) {
            const int kb = st * 32;

            // ---- S = K^T Q: 32 keys x 32 queries = 4 C tiles, 8 WMMAs.
            // Each K fragment pair feeds both q-tiles (4 WMMAs per 2 frag loads).
            v8f s[2][2];  // [qtile][keytile]
#pragma unroll
            for (int kt2 = 0; kt2 < 2; ++kt2) {
                const _Float16* r = sK + (kb + kt2 * 16 + col) * LSTR;
                const v16h a0 = load_a_frag(r, hi);        // d = 0..31
                const v16h a1 = load_a_frag(r + 32, hi);   // d = 32..63
#pragma unroll
                for (int t = 0; t < 2; ++t) {
                    v8f acc = __builtin_amdgcn_wmma_f32_16x16x32_f16(
                        false, a0, false, qf[t][0], (short)0, v8f{}, false, false);
                    acc = __builtin_amdgcn_wmma_f32_16x16x32_f16(
                        false, a1, false, qf[t][1], (short)0, acc, false, false);
                    s[t][kt2] = acc;
                }
            }

            // ---- Online softmax per q-tile over these 32 keys
            v16h pb[2];
#pragma unroll
            for (int t = 0; t < 2; ++t) {
                const v8f s0 = s[t][0], s1 = s[t][1];
                float tmax = fmaxf(s0[0], s1[0]);
#pragma unroll
                for (int i = 1; i < 8; ++i) tmax = fmaxf(tmax, fmaxf(s0[i], s1[i]));
                tmax = fmaxf(tmax, __shfl_xor(tmax, 16, 32));
                const float mn   = fmaxf(m[t], tmax);
                const float corr = __expf(m[t] - mn);
                m[t] = mn;

                float p0[8], p1[8], rsum = 0.0f;
#pragma unroll
                for (int i = 0; i < 8; ++i) {
                    p0[i] = __expf(s0[i] - mn); rsum += p0[i];
                    p1[i] = __expf(s1[i] - mn); rsum += p1[i];
                }
                rsum += __shfl_xor(rsum, 16, 32);
                l[t] = l[t] * corr + rsum;
#pragma unroll
                for (int c = 0; c < 4; ++c)
#pragma unroll
                    for (int i = 0; i < 8; ++i) o[t][c][i] *= corr;

                // Repack P (C layout) into 32x16 B-fragment layout:
                // lanes 0-15 need all 16 rows of key-tile0, lanes 16-31 of key-tile1.
#pragma unroll
                for (int i = 0; i < 8; ++i) {
                    const float x0 = __shfl_xor(p0[i], 16, 32);
                    const float x1 = __shfl_xor(p1[i], 16, 32);
                    pb[t][i]     = (_Float16)(hi ? x1 : p0[i]);  // K = 16*hi + i
                    pb[t][i + 8] = (_Float16)(hi ? p1[i] : x0);  // K = 16*hi + 8 + i
                }
            }

            // ---- O += V * P: each V fragment feeds both q-tiles
#pragma unroll
            for (int c = 0; c < 4; ++c) {
                const v16h vf = load_a_frag(sV + (c * 16 + col) * LSTR + kb, hi);
#pragma unroll
                for (int t = 0; t < 2; ++t)
                    o[t][c] = __builtin_amdgcn_wmma_f32_16x16x32_f16(
                        false, vf, false, pb[t], (short)0, o[t][c], false, false);
            }
        }
    }

    // ---- Epilogue: R = O / l to rows 0..63; Q passthrough to rows 64..127
    float* Ob = Og + (size_t)b * (2 * D_DIM) * S_LEN;
#pragma unroll
    for (int t = 0; t < 2; ++t) {
        const float invl = 1.0f / l[t];
        const int q = q0 + wid * 32 + t * 16 + col;
#pragma unroll
        for (int c = 0; c < 4; ++c)
#pragma unroll
            for (int i = 0; i < 8; ++i) {
                const int d = c * 16 + hi * 8 + i;  // C layout: VGPR i -> M = i + 8*hi
                Ob[(size_t)d * S_LEN + q] = o[t][c][i] * invl;
            }
    }
#pragma unroll
    for (int e = tid; e < QT * D_DIM; e += NTHREADS) {
        const int d = e >> 8, qq = e & 255;
        Ob[(size_t)(D_DIM + d) * S_LEN + q0 + qq] = Qb[(size_t)d * S_LEN + q0 + qq];
    }
}

extern "C" void kernel_launch(void* const* d_in, const int* in_sizes, int n_in,
                              void* d_out, int out_size, void* d_ws, size_t ws_size,
                              hipStream_t stream) {
    (void)in_sizes; (void)n_in; (void)out_size; (void)d_ws; (void)ws_size;
    const float* K = (const float*)d_in[0];
    const float* V = (const float*)d_in[1];
    const float* Q = (const float*)d_in[2];
    float* out = (float*)d_out;
    dim3 grid(S_LEN / QT, 4 /*B*/);
    fa_fwd_kernel<<<grid, NTHREADS, 0, stream>>>(K, V, Q, out);
}
